// Attention_8693013807599
// MI455X (gfx1250) — compile-verified, matrix operand order checked by reading, not run on hardware
//
#include <hip/hip_runtime.h>

#define B_   8
#define N_   2048
#define DIN_ 1024
#define H_   4
#define DH_  256
#define HD_  1024   // H_*DH_
#define DOUT_ 256

typedef __attribute__((ext_vector_type(16))) _Float16 v16h;
typedef __attribute__((ext_vector_type(8)))  float    v8f;

__device__ __forceinline__ v8f wmma_f16(v16h a, v16h b, v8f c) {
  // D = A(16x32 f16) x B(32x16 f16) + C(16x16 f32)
  return __builtin_amdgcn_wmma_f32_16x16x32_f16(
      /*neg_a=*/false, a, /*neg_b=*/false, b,
      /*c_mod=*/(short)0, c, /*reuse_a=*/false, /*reuse_b=*/false);
}

__device__ __forceinline__ v8f vzero8() {
  v8f z = {0.0f, 0.0f, 0.0f, 0.0f, 0.0f, 0.0f, 0.0f, 0.0f};
  return z;
}

// CDNA5 async global->LDS copy (16B per lane), tracked by ASYNCcnt.
// LDS VGPR address = low 32 bits of the generic pointer (ISA 10.2 aperture rule).
__device__ __forceinline__ void async_copy_b128(void* lds, const void* gsrc) {
#if defined(__gfx1250__)
  asm volatile("global_load_async_to_lds_b128 %0, %1, off"
               :: "v"((unsigned int)(size_t)lds), "v"(gsrc)
               : "memory");
#else
  *(uint4*)lds = *(const uint4*)gsrc;
#endif
}

__device__ __forceinline__ void wait_async0() {
#if defined(__gfx1250__)
  asm volatile("s_wait_asynccnt 0" ::: "memory");
#endif
}

// ---------------------------------------------------------------------------
// f32 -> f16 cast (grid-stride)
// ---------------------------------------------------------------------------
__global__ void cast_f32_f16(const float* __restrict__ s, _Float16* __restrict__ d, int n) {
  int i = blockIdx.x * blockDim.x + threadIdx.x;
  int stride = gridDim.x * blockDim.x;
  for (; i < n; i += stride) d[i] = (_Float16)s[i];
}

// ---------------------------------------------------------------------------
// WMMA GEMM: out[M,Nc] = A[M,K](f16) * W[K,Nc](f16) + bias
// Block tile 128x128, 256 threads = 8 waves in 4x2; each wave 32x64 (2x4 WMMA tiles).
// A tile staged via async global->LDS; B tile transposed via VGPR scatter.
// MODE 0: store f16 scattered into [B,H,N,DH] (QKV path)
// MODE 1: store f32 row-major (output projection)
// ---------------------------------------------------------------------------
template <int MODE>
__global__ __launch_bounds__(256)
void wmma_gemm(const _Float16* __restrict__ A, const _Float16* __restrict__ W,
               const float* __restrict__ bias, void* __restrict__ outp,
               int M, int K, int Nc) {
  __shared__ alignas(32) _Float16 As[128][48];  // 128 rows x 32 K-halves (+16 pad)
  __shared__ alignas(32) _Float16 Bs[128][48];  // 128 cols x 32 K-halves (transposed)

  const int tid  = threadIdx.x;
  const int wave = tid >> 5, lane = tid & 31;
  const int wm = wave >> 1, wn = wave & 1;
  const int r16 = lane & 15, hi = lane >> 4;
  const int m0 = blockIdx.x * 128, n0 = blockIdx.y * 128;

  v8f acc[2][4];
#pragma unroll
  for (int t = 0; t < 2; ++t)
#pragma unroll
    for (int u = 0; u < 4; ++u) acc[t][u] = vzero8();

  const int kTiles = K >> 5;
  for (int kc = 0; kc < kTiles; ++kc) {
    const int k0 = kc << 5;
    __syncthreads();  // all waves done reading previous tiles
    // Stage A tile (row-major) via async DMA: 128 rows x 32 halves = 512 x b128
    for (int idx = tid; idx < 512; idx += 256) {
      int row = idx >> 2, seg = idx & 3;
      async_copy_b128(&As[row][seg * 8],
                      &A[(size_t)(m0 + row) * K + k0 + seg * 8]);
    }
    // Stage B tile transposed: read W rows contiguously, scatter to [col][k]
    for (int idx = tid; idx < 512; idx += 256) {
      int k = idx >> 4, cs = idx & 15;
      uint4 dvec = *(const uint4*)&W[(size_t)(k0 + k) * Nc + n0 + cs * 8];
      const _Float16* hp = (const _Float16*)&dvec;
#pragma unroll
      for (int j = 0; j < 8; ++j) Bs[cs * 8 + j][k] = hp[j];
      if (kc + 1 < kTiles)
        __builtin_prefetch(&W[(size_t)(k0 + 32 + k) * Nc + n0 + cs * 8], 0, 0);
    }
    wait_async0();   // this wave's async LDS writes landed
    __syncthreads(); // tile visible to all waves

    v16h af[2], bf[4];
#pragma unroll
    for (int t = 0; t < 2; ++t)
      af[t] = *(const v16h*)&As[wm * 32 + t * 16 + r16][hi * 16];
#pragma unroll
    for (int u = 0; u < 4; ++u)
      bf[u] = *(const v16h*)&Bs[wn * 64 + u * 16 + r16][hi * 16];
#pragma unroll
    for (int t = 0; t < 2; ++t)
#pragma unroll
      for (int u = 0; u < 4; ++u) acc[t][u] = wmma_f16(af[t], bf[u], acc[t][u]);
  }

  // Epilogue: C layout -> row m = r + 8*hi, col n = lane%16 (per 16x16 tile)
#pragma unroll
  for (int t = 0; t < 2; ++t)
#pragma unroll
    for (int u = 0; u < 4; ++u)
#pragma unroll
      for (int r = 0; r < 8; ++r) {
        int m = m0 + wm * 32 + t * 16 + r + 8 * hi;
        int n = n0 + wn * 64 + u * 16 + r16;
        float val = acc[t][u][r] + bias[n];
        if (MODE == 0) {
          int h = n >> 8, dh = n & 255;     // n in [0,1024): head, head-dim
          int b = m >> 11, nn = m & 2047;   // m in [0,16384): batch, seq
          ((_Float16*)outp)[(((size_t)(b * H_ + h) * N_ + nn) << 8) + dh] = (_Float16)val;
        } else {
          ((float*)outp)[(size_t)m * Nc + n] = val;
        }
      }
}

// ---------------------------------------------------------------------------
// Flash attention: one block per (b,h) x 64 query rows; 4 waves x 16 rows.
// Q held in registers (8 A-fragments), K staged via async global->LDS,
// V staged transposed via VGPR scatter; online softmax on the C layout.
// ---------------------------------------------------------------------------
__global__ __launch_bounds__(128)
void attn_kernel(const _Float16* __restrict__ Q, const _Float16* __restrict__ Km,
                 const _Float16* __restrict__ V, _Float16* __restrict__ ctx) {
  __shared__ alignas(32) _Float16 Ks[32][DH_ + 16];  // 32 keys x 256 dh (+pad)
  __shared__ alignas(32) _Float16 Vt[DH_][48];       // 256 dh x 32 keys (+pad)
  __shared__ alignas(32) _Float16 Ps[4][16][48];     // per-wave P relayout buffer

  const int bh = blockIdx.x;
  const int b = bh >> 2, h = bh & 3;
  const int tid = threadIdx.x;
  const int wave = tid >> 5, lane = tid & 31;
  const int r16 = lane & 15, hi = lane >> 4;
  const int qBase = blockIdx.y * 64 + wave * 16;

  const _Float16* Qp = Q + (size_t)bh * N_ * DH_;
  const _Float16* Kp = Km + (size_t)bh * N_ * DH_;
  const _Float16* Vp = V + (size_t)bh * N_ * DH_;

  // Preload 16 query rows as 8 A-fragments (16x32 each over DH=256)
  v16h qf[8];
#pragma unroll
  for (int c = 0; c < 8; ++c)
    qf[c] = *(const v16h*)&Qp[(size_t)(qBase + r16) * DH_ + c * 32 + hi * 16];

  v8f o[16];
#pragma unroll
  for (int t = 0; t < 16; ++t) o[t] = vzero8();
  v8f mrow, lrow;
#pragma unroll
  for (int r = 0; r < 8; ++r) { mrow[r] = -1e30f; lrow[r] = 0.0f; }

  for (int kt = 0; kt < N_ / 32; ++kt) {
    const int keyBase = kt * 32;
    __syncthreads();  // all waves done reading previous K/V tiles
    // K tile row-major via async DMA (rows = keys, contiguous along dh)
    for (int idx = tid; idx < 32 * 32; idx += 128) {
      int r = idx >> 5, seg = idx & 31;
      async_copy_b128(&Ks[r][seg * 8],
                      &Kp[(size_t)(keyBase + r) * DH_ + seg * 8]);
    }
    // V tile transposed (rows = dh, contiguous along keys -> B-frag of P*V)
    for (int idx = tid; idx < 32 * 32; idx += 128) {
      int key = idx >> 5, seg = idx & 31;
      uint4 dvec = *(const uint4*)&Vp[(size_t)(keyBase + key) * DH_ + seg * 8];
      const _Float16* hp = (const _Float16*)&dvec;
#pragma unroll
      for (int j = 0; j < 8; ++j) Vt[seg * 8 + j][key] = hp[j];
    }
    wait_async0();
    __syncthreads();

    // S = Q * K^T for 32 keys (two 16x16 tiles), K-dim 256 in 8 chunks
    v8f s0 = vzero8(), s1 = vzero8();
#pragma unroll
    for (int c = 0; c < 8; ++c) {
      v16h kb0 = *(const v16h*)&Ks[r16][c * 32 + hi * 16];
      v16h kb1 = *(const v16h*)&Ks[16 + r16][c * 32 + hi * 16];
      s0 = wmma_f16(qf[c], kb0, s0);
      s1 = wmma_f16(qf[c], kb1, s1);
    }
    const float scale = 0.0625f;  // 1/sqrt(256)
    s0 = s0 * scale;
    s1 = s1 * scale;

    // Row max across 16 lanes (rows live at r + 8*hi; cols across lanes 0..15)
    v8f tmax;
#pragma unroll
    for (int r = 0; r < 8; ++r) tmax[r] = fmaxf(s0[r], s1[r]);
#pragma unroll
    for (int msk = 1; msk < 16; msk <<= 1)
#pragma unroll
      for (int r = 0; r < 8; ++r)
        tmax[r] = fmaxf(tmax[r], __shfl_xor(tmax[r], msk, 32));

    v8f mnew, alpha;
#pragma unroll
    for (int r = 0; r < 8; ++r) {
      mnew[r]  = fmaxf(mrow[r], tmax[r]);
      alpha[r] = __expf(mrow[r] - mnew[r]);
      mrow[r]  = mnew[r];
    }
#pragma unroll
    for (int r = 0; r < 8; ++r) {
      s0[r] = __expf(s0[r] - mnew[r]);
      s1[r] = __expf(s1[r] - mnew[r]);
    }
    v8f rsum;
#pragma unroll
    for (int r = 0; r < 8; ++r) rsum[r] = s0[r] + s1[r];
#pragma unroll
    for (int msk = 1; msk < 16; msk <<= 1)
#pragma unroll
      for (int r = 0; r < 8; ++r) rsum[r] += __shfl_xor(rsum[r], msk, 32);
#pragma unroll
    for (int r = 0; r < 8; ++r) lrow[r] = lrow[r] * alpha[r] + rsum[r];
#pragma unroll
    for (int t = 0; t < 16; ++t) o[t] = o[t] * alpha;

    // Relayout P (C layout: col-per-lane) -> A-fragment (row-per-lane) via LDS
#pragma unroll
    for (int r = 0; r < 8; ++r) {
      int prow = r + 8 * hi;
      Ps[wave][prow][r16]      = (_Float16)s0[r];
      Ps[wave][prow][16 + r16] = (_Float16)s1[r];
    }
    asm volatile("s_wait_dscnt 0" ::: "memory");  // intra-wave LDS RAW
    v16h pf = *(const v16h*)&Ps[wave][r16][hi * 16];

    // O += P(16x32) * V(32x256) as 16 WMMAs over dh chunks
#pragma unroll
    for (int t = 0; t < 16; ++t) {
      v16h vb = *(const v16h*)&Vt[t * 16 + r16][hi * 16];
      o[t] = wmma_f16(pf, vb, o[t]);
    }
  }

  // Normalize and store context in [B, N, H*DH] (f16)
  v8f inv;
#pragma unroll
  for (int r = 0; r < 8; ++r) inv[r] = 1.0f / lrow[r];
#pragma unroll
  for (int t = 0; t < 16; ++t)
#pragma unroll
    for (int r = 0; r < 8; ++r) {
      int n  = qBase + r + 8 * hi;
      int dh = t * 16 + r16;
      ctx[((size_t)b * N_ + n) * HD_ + h * DH_ + dh] = (_Float16)(o[t][r] * inv[r]);
    }
}

// ---------------------------------------------------------------------------
// Launch
// ---------------------------------------------------------------------------
extern "C" void kernel_launch(void* const* d_in, const int* in_sizes, int n_in,
                              void* d_out, int out_size, void* d_ws, size_t ws_size,
                              hipStream_t stream) {
  const float* x  = (const float*)d_in[0];
  const float* Wq = (const float*)d_in[1];
  const float* bq = (const float*)d_in[2];
  const float* Wk = (const float*)d_in[3];
  const float* bk = (const float*)d_in[4];
  const float* Wv = (const float*)d_in[5];
  const float* bv = (const float*)d_in[6];
  const float* Wo = (const float*)d_in[7];
  const float* bo = (const float*)d_in[8];
  float* out = (float*)d_out;

  char* ws = (char*)d_ws;
  size_t off = 0;
  auto take = [&](size_t elems) {
    _Float16* p = (_Float16*)(ws + off);
    off += ((elems * sizeof(_Float16)) + 255) & ~(size_t)255;
    return p;
  };
  _Float16* xh  = take((size_t)B_ * N_ * DIN_);       // 32 MB
  _Float16* wqh = take((size_t)DIN_ * HD_);
  _Float16* wkh = take((size_t)DIN_ * HD_);
  _Float16* wvh = take((size_t)DIN_ * HD_);
  _Float16* woh = take((size_t)HD_ * DOUT_);
  _Float16* qws = take((size_t)B_ * H_ * N_ * DH_);   // 32 MB each
  _Float16* kws = take((size_t)B_ * H_ * N_ * DH_);
  _Float16* vws = take((size_t)B_ * H_ * N_ * DH_);
  _Float16* ctx = take((size_t)B_ * N_ * HD_);        // 32 MB

  // 1) f32 -> f16 casts
  cast_f32_f16<<<4096, 256, 0, stream>>>(x,  xh,  B_ * N_ * DIN_);
  cast_f32_f16<<<1024, 256, 0, stream>>>(Wq, wqh, DIN_ * HD_);
  cast_f32_f16<<<1024, 256, 0, stream>>>(Wk, wkh, DIN_ * HD_);
  cast_f32_f16<<<1024, 256, 0, stream>>>(Wv, wvh, DIN_ * HD_);
  cast_f32_f16<<<256,  256, 0, stream>>>(Wo, woh, HD_ * DOUT_);

  // 2) QKV projections: [16384,1024] x [1024,1024] -> f16 [B,H,N,DH]
  const int M = B_ * N_;
  dim3 gQKV(M / 128, HD_ / 128);
  wmma_gemm<0><<<gQKV, 256, 0, stream>>>(xh, wqh, bq, qws, M, DIN_, HD_);
  wmma_gemm<0><<<gQKV, 256, 0, stream>>>(xh, wkh, bk, kws, M, DIN_, HD_);
  wmma_gemm<0><<<gQKV, 256, 0, stream>>>(xh, wvh, bv, vws, M, DIN_, HD_);

  // 3) Flash attention: 32 (b,h) blocks x 32 query tiles of 64 rows
  dim3 gAttn(B_ * H_, N_ / 64);
  attn_kernel<<<gAttn, 128, 0, stream>>>(qws, kws, vws, ctx);

  // 4) Output projection: [16384,1024] x [1024,256] + bo -> f32 out
  dim3 gOut(M / 128, DOUT_ / 128);
  wmma_gemm<1><<<gOut, 256, 0, stream>>>(ctx, woh, bo, out, M, HD_, DOUT_);
}
